// GNNEncoder_73383811219652
// MI455X (gfx1250) — matrix operand order repaired and checked by reading, not compiled, not run on hardware
//
#include <hip/hip_runtime.h>
#include <hip/hip_bf16.h>

#define B_   4
#define N_   2048
#define KNN  20
#define EPSV 1e-5f
#define PTS_PAD 32

typedef __attribute__((ext_vector_type(16))) _Float16 v16h;
typedef __attribute__((ext_vector_type(8)))  _Float16 v8h;
typedef __attribute__((ext_vector_type(8)))  float    v8f;

union Frag16 { v16h v; v8h h[2]; };

// ---------------------------------------------------------------- utilities
__global__ void zero_f32(float* p, int n) {
    int t = blockIdx.x * blockDim.x + threadIdx.x;
    if (t < n) p[t] = 0.0f;
}

// points (B,N,3) f32 -> xh (B,N,PTS_PAD) f16, zero padded
__global__ void pad_points_f16(const float* __restrict__ pts, _Float16* __restrict__ xh) {
    int t = blockIdx.x * blockDim.x + threadIdx.x;
    if (t >= B_ * N_ * PTS_PAD) return;
    int ch = t & (PTS_PAD - 1);
    int n = t >> 5;
    xh[t] = (ch < 3) ? (_Float16)pts[(size_t)n * 3 + ch] : (_Float16)0.0f;
}

// W (2c, h) f32 -> Wht (h, 2*cpad) f16, transposed, zero-padded per half
__global__ void wht_convert(const float* __restrict__ W, _Float16* __restrict__ Wht,
                            int c, int cpad, int h) {
    int Kpad = 2 * cpad;
    int t = blockIdx.x * blockDim.x + threadIdx.x;
    if (t >= h * Kpad) return;
    int col = t / Kpad, kk = t % Kpad;
    float v = 0.0f;
    if (kk < cpad) { if (kk < c) v = W[(size_t)kk * h + col]; }            // x_i half
    else { int r = kk - cpad; if (r < c) v = W[(size_t)(c + r) * h + col]; } // (x_j-x_i) half
    Wht[t] = (_Float16)v;
}

__global__ void sqnorm_kernel(const float* __restrict__ x, float* __restrict__ sq, int c) {
    int t = blockIdx.x * blockDim.x + threadIdx.x; // over B*N
    if (t >= B_ * N_) return;
    const float* r = x + (size_t)t * c;
    float s = 0.0f;
    for (int i = 0; i < c; ++i) s += r[i] * r[i];
    sq[t] = s;
}

// ------------------------------------------------- Gram / distance via WMMA
// One wave per 16x16 tile of D[b]:  D = sq_i + sq_j - 2 * x_i . x_j
// cpad is a multiple of 32 or of 16 with loop-covering zeros; all loads unconditional.
__global__ void gram_dist_wmma(const _Float16* __restrict__ xh,
                               const float* __restrict__ sq,
                               float* __restrict__ D, int cpad) {
    int it = blockIdx.x, jt = blockIdx.y, b = blockIdx.z;
    int lane = threadIdx.x & 31;
    int hi = lane >> 4, lo = lane & 15;
    const _Float16* xb = xh + (size_t)b * N_ * cpad;
    const _Float16* arow = xb + (size_t)(it * 16 + lo) * cpad;
    const _Float16* brow = xb + (size_t)(jt * 16 + lo) * cpad;
    v8f acc = {};
    for (int kb = 0; kb < cpad; kb += 32) {
        Frag16 A, Bf;
        A.h[0] = *(const v8h*)(arow + kb + hi * 8);
        A.h[1] = *(const v8h*)(arow + kb + 16 + hi * 8);
        Bf.h[0] = *(const v8h*)(brow + kb + hi * 8);
        Bf.h[1] = *(const v8h*)(brow + kb + 16 + hi * 8);
        acc = __builtin_amdgcn_wmma_f32_16x16x32_f16(false, A.v, false, Bf.v,
                                                     (short)0, acc, false, false);
    }
    const float* sqb = sq + b * N_;
    float sj = sqb[jt * 16 + lo];
#pragma unroll
    for (int r = 0; r < 8; ++r) {
        int i = it * 16 + (hi << 3) + r;
        int j = jt * 16 + lo;
        float d = sqb[i] + sj - 2.0f * acc[r];
        if (i == j) d = 1e30f;
        D[((size_t)b * N_ + i) * N_ + j] = d;
    }
}

// ---------------------------------------------------- top-K (one wave/row)
__global__ void topk_kernel(const float* __restrict__ D, int* __restrict__ idx) {
    int row = blockIdx.x;                  // over B*N
    int lane = threadIdx.x & 31;
    const float* dr = D + (size_t)row * N_;
    float vals[KNN]; int ids[KNN];
#pragma unroll
    for (int t = 0; t < KNN; ++t) { vals[t] = 1e30f; ids[t] = 0; }
    for (int j = lane; j < N_; j += 32) {
        float cv = dr[j]; int ci = j;
        if (cv < vals[KNN - 1]) {
#pragma unroll
            for (int p = 0; p < KNN; ++p) {
                bool less = cv < vals[p];
                float tv = vals[p]; int ti = ids[p];
                vals[p] = less ? cv : tv;  ids[p] = less ? ci : ti;
                cv      = less ? tv : cv;  ci      = less ? ti : ci;
            }
        }
    }
    for (int t = 0; t < KNN; ++t) {
        float bv = vals[0]; int bid = ids[0]; int blane = lane;
        for (int off = 16; off > 0; off >>= 1) {
            float ov = __shfl_xor(bv, off);
            int  oid = __shfl_xor(bid, off);
            int  ol  = __shfl_xor(blane, off);
            if (ov < bv || (ov == bv && ol < blane)) { bv = ov; bid = oid; blane = ol; }
        }
        if (lane == 0) idx[(size_t)row * KNN + t] = bid;
        if (lane == blane) {
#pragma unroll
            for (int p = 0; p < KNN - 1; ++p) { vals[p] = vals[p + 1]; ids[p] = ids[p + 1]; }
            vals[KNN - 1] = 1e30f; ids[KNN - 1] = 0;
        }
    }
}

// ------------------------------------------------------ EdgeConv via WMMA
// Concatenated feature [x_i | x_j - x_i], each half padded to cpad.
// One 8-element span never straddles the half boundary (spans start at mult. of 8,
// cpad is a multiple of 16), so region selection is one uniform branch per span.
__device__ inline v8h load_feat8(const _Float16* __restrict__ xn,
                                 const _Float16* __restrict__ xj, int s, int cpad) {
    if (s < cpad) return *(const v8h*)(xn + s);
    v8h a = *(const v8h*)(xj + (s - cpad));
    v8h b = *(const v8h*)(xn + (s - cpad));
    return a - b;
}

// 16x32 output per wave (2 column tiles share the gathered A fragment)
template <int WRITE>
__global__ void edgeconv_wmma(const _Float16* __restrict__ xh,
                              const int* __restrict__ idx,
                              const _Float16* __restrict__ Wht,
                              const float* __restrict__ bias,
                              const float* __restrict__ scale,
                              const float* __restrict__ shift,
                              float* __restrict__ ssum, float* __restrict__ ssq,
                              unsigned int* __restrict__ xnew,
                              int cpad, int h) {
    int rt = blockIdx.x, ct = blockIdx.y, b = blockIdx.z;
    int lane = threadIdx.x & 31;
    int hi = lane >> 4, lo = lane & 15;
    int Kpad = 2 * cpad;
    const _Float16* xb = xh + (size_t)b * N_ * cpad;
    int m = rt * 16 + lo;                    // edge row within batch
    int n = m / KNN, kk = m % KNN;
    int j = idx[((size_t)b * N_ + n) * KNN + kk];
    const _Float16* xn = xb + (size_t)n * cpad;
    const _Float16* xj = xb + (size_t)j * cpad;
    int col0 = ct * 32 + lo;
    const _Float16* w0 = Wht + (size_t)col0 * Kpad;
    const _Float16* w1 = w0 + (size_t)16 * Kpad;
    v8f acc0 = {}, acc1 = {};
    for (int kb = 0; kb < Kpad; kb += 32) {
        Frag16 A, B0, B1;
        A.h[0] = load_feat8(xn, xj, kb + hi * 8, cpad);
        A.h[1] = load_feat8(xn, xj, kb + 16 + hi * 8, cpad);
        int bo = kb + hi * 8;
        B0.h[0] = *(const v8h*)(w0 + bo);
        B0.h[1] = *(const v8h*)(w0 + bo + 16);
        B1.h[0] = *(const v8h*)(w1 + bo);
        B1.h[1] = *(const v8h*)(w1 + bo + 16);
        acc0 = __builtin_amdgcn_wmma_f32_16x16x32_f16(false, A.v, false, B0.v,
                                                      (short)0, acc0, false, false);
        acc1 = __builtin_amdgcn_wmma_f32_16x16x32_f16(false, A.v, false, B1.v,
                                                      (short)0, acc1, false, false);
    }
    float bc0 = bias[col0], bc1 = bias[col0 + 16];
    if (WRITE == 0) {
        float s0 = 0.f, q0 = 0.f, s1 = 0.f, q1 = 0.f;
#pragma unroll
        for (int r = 0; r < 8; ++r) {
            float v0 = acc0[r] + bc0; s0 += v0; q0 += v0 * v0;
            float v1 = acc1[r] + bc1; s1 += v1; q1 += v1 * v1;
        }
        s0 += __shfl_down(s0, 16); q0 += __shfl_down(q0, 16);
        s1 += __shfl_down(s1, 16); q1 += __shfl_down(q1, 16);
        if (hi == 0) {
            atomicAdd(&ssum[col0], s0);      atomicAdd(&ssq[col0], q0);
            atomicAdd(&ssum[col0 + 16], s1); atomicAdd(&ssq[col0 + 16], q1);
        }
    } else {
        float sc0 = scale[col0], sh0 = shift[col0];
        float sc1 = scale[col0 + 16], sh1 = shift[col0 + 16];
#pragma unroll
        for (int r = 0; r < 8; ++r) {
            int mr = rt * 16 + (hi << 3) + r;
            int nr = mr / KNN;
            size_t base = ((size_t)b * N_ + nr) * h;
            float y0 = fmaxf((acc0[r] + bc0) * sc0 + sh0, 0.0f);
            float y1 = fmaxf((acc1[r] + bc1) * sc1 + sh1, 0.0f);
            atomicMax(&xnew[base + col0],      __float_as_uint(y0));
            atomicMax(&xnew[base + col0 + 16], __float_as_uint(y1));
        }
    }
}

// ------------------------------------------------------------- batch-norms
__global__ void chan_stats(const float* __restrict__ x, float* __restrict__ sum,
                           float* __restrict__ sumsq, int rows, int h) {
    int ch = blockIdx.x;
    float s = 0.0f, q = 0.0f;
    for (int r = threadIdx.x; r < rows; r += blockDim.x) {
        float v = x[(size_t)r * h + ch];
        s += v; q += v * v;
    }
    __shared__ float rs[256], rq[256];
    rs[threadIdx.x] = s; rq[threadIdx.x] = q;
    __syncthreads();
    for (int off = 128; off > 0; off >>= 1) {
        if ((int)threadIdx.x < off) { rs[threadIdx.x] += rs[threadIdx.x + off];
                                      rq[threadIdx.x] += rq[threadIdx.x + off]; }
        __syncthreads();
    }
    if (threadIdx.x == 0) { sum[ch] = rs[0]; sumsq[ch] = rq[0]; }
}

__global__ void finalize_stats(const float* __restrict__ sum, const float* __restrict__ sumsq,
                               const float* __restrict__ g, const float* __restrict__ beta,
                               float inv_cnt, float* __restrict__ scale,
                               float* __restrict__ shift, int h) {
    int ch = blockIdx.x * blockDim.x + threadIdx.x;
    if (ch >= h) return;
    float m = sum[ch] * inv_cnt;
    float v = sumsq[ch] * inv_cnt - m * m;
    float sc = g[ch] * rsqrtf(v + EPSV);
    scale[ch] = sc;
    shift[ch] = beta[ch] - m * sc;
}

__global__ void bn_apply_f16(const float* __restrict__ x, const float* __restrict__ scale,
                             const float* __restrict__ shift, float* __restrict__ xout,
                             _Float16* __restrict__ xh, int total, int h) {
    int t = blockIdx.x * blockDim.x + threadIdx.x;
    if (t >= total) return;
    int ch = t % h;
    float y = fmaxf(x[t] * scale[ch] + shift[ch], 0.0f);
    xout[t] = y;
    xh[t] = (_Float16)y;
}

// ------------------------------------------------------------------- head
__global__ void mean_pool(const float* __restrict__ x, float* __restrict__ pooled, int h) {
    int t = blockIdx.x * blockDim.x + threadIdx.x;  // over B*h
    if (t >= B_ * h) return;
    int b = t / h, ch = t % h;
    float s = 0.0f;
    for (int n = 0; n < N_; ++n) s += x[((size_t)b * N_ + n) * h + ch];
    pooled[t] = s * (1.0f / N_);
}

__global__ void fc_bn(const float* __restrict__ in, const float* __restrict__ W,
                      const float* __restrict__ bias, const float* __restrict__ g,
                      const float* __restrict__ beta, float* __restrict__ out,
                      int cin, int cout, int do_relu) {
    int col = blockIdx.x * blockDim.x + threadIdx.x;
    if (col >= cout) return;
    float a0 = 0, a1 = 0, a2 = 0, a3 = 0;
    for (int k = 0; k < cin; ++k) {
        float w = W[(size_t)k * cout + col];
        a0 += in[0 * cin + k] * w; a1 += in[1 * cin + k] * w;
        a2 += in[2 * cin + k] * w; a3 += in[3 * cin + k] * w;
    }
    float bb = bias[col];
    a0 += bb; a1 += bb; a2 += bb; a3 += bb;
    float m = 0.25f * (a0 + a1 + a2 + a3);
    float v = 0.25f * ((a0 - m) * (a0 - m) + (a1 - m) * (a1 - m) +
                       (a2 - m) * (a2 - m) + (a3 - m) * (a3 - m));
    float sc = g[col] * rsqrtf(v + EPSV);
    float sh = beta[col] - m * sc;
    float y0 = a0 * sc + sh, y1 = a1 * sc + sh, y2 = a2 * sc + sh, y3 = a3 * sc + sh;
    if (do_relu) { y0 = fmaxf(y0, 0.f); y1 = fmaxf(y1, 0.f); y2 = fmaxf(y2, 0.f); y3 = fmaxf(y3, 0.f); }
    out[0 * cout + col] = y0; out[1 * cout + col] = y1;
    out[2 * cout + col] = y2; out[3 * cout + col] = y3;
}

// ------------------------------------------------------------------ launch
extern "C" void kernel_launch(void* const* d_in, const int* in_sizes, int n_in,
                              void* d_out, int out_size, void* d_ws, size_t ws_size,
                              hipStream_t stream) {
    const float* points = (const float*)d_in[0];

    char* w = (char*)d_ws;
    size_t off = 0;
    auto take = [&](size_t bytes) -> void* {
        void* p = w + off; off += (bytes + 255) & ~(size_t)255; return p;
    };
    float*     D     = (float*)take((size_t)B_ * N_ * N_ * 4);
    int*       idxb  = (int*)take((size_t)B_ * N_ * KNN * 4);
    float*     x0    = (float*)take((size_t)B_ * N_ * 512 * 4);
    float*     x1    = (float*)take((size_t)B_ * N_ * 512 * 4);
    float*     xP    = (float*)take((size_t)B_ * N_ * 512 * 4);   // pre-BN2 (atomicMax target)
    _Float16*  xh    = (_Float16*)take((size_t)B_ * N_ * 512 * 2);
    _Float16*  Wht   = (_Float16*)take((size_t)512 * 1024 * 2);   // (h, Kpad) transposed
    float*     sq    = (float*)take((size_t)B_ * N_ * 4);
    float*     s1    = (float*)take(512 * 4);
    float*     q1    = (float*)take(512 * 4);
    float*     sc1   = (float*)take(512 * 4);
    float*     sh1   = (float*)take(512 * 4);
    float*     s2    = (float*)take(512 * 4);
    float*     q2    = (float*)take(512 * 4);
    float*     sc2   = (float*)take(512 * 4);
    float*     sh2   = (float*)take(512 * 4);
    float*     pooled= (float*)take((size_t)B_ * 512 * 4);
    float*     hdd   = (float*)take((size_t)B_ * 512 * 4);

    const int HID[4] = {64, 128, 256, 512};
    const float* xcur = points;
    int c = 3;
    int cpad = PTS_PAD;   // f16 feature row stride (padded)

    {   // initial padded f16 copy of points
        int n = B_ * N_ * PTS_PAD;
        pad_points_f16<<<(n + 255) / 256, 256, 0, stream>>>(points, xh);
    }

    float* outbufs[2] = {x0, x1};
    for (int l = 0; l < 4; ++l) {
        int h = HID[l];
        const float* Wl = (const float*)d_in[1 + l * 6 + 0];
        const float* bl = (const float*)d_in[1 + l * 6 + 1];
        const float* eg = (const float*)d_in[1 + l * 6 + 2];
        const float* eb = (const float*)d_in[1 + l * 6 + 3];
        const float* ng = (const float*)d_in[1 + l * 6 + 4];
        const float* nb = (const float*)d_in[1 + l * 6 + 5];

        // --- kNN on current features ---
        sqnorm_kernel<<<(B_ * N_ + 255) / 256, 256, 0, stream>>>(xcur, sq, c);
        gram_dist_wmma<<<dim3(N_ / 16, N_ / 16, B_), 32, 0, stream>>>(xh, sq, D, cpad);
        topk_kernel<<<B_ * N_, 32, 0, stream>>>(D, idxb);

        // --- weights: transpose + pad to f16 ---
        {
            int n = h * 2 * cpad;
            wht_convert<<<(n + 255) / 256, 256, 0, stream>>>(Wl, Wht, c, cpad, h);
        }

        // --- EdgeConv pass A: BN1 stats ---
        zero_f32<<<(h + 255) / 256, 256, 0, stream>>>(s1, h);
        zero_f32<<<(h + 255) / 256, 256, 0, stream>>>(q1, h);
        {
            int n = B_ * N_ * h;
            zero_f32<<<(n + 255) / 256, 256, 0, stream>>>(xP, n); // atomicMax base (relu floor)
        }
        dim3 egrid((N_ * KNN) / 16, h / 32, B_);
        edgeconv_wmma<0><<<egrid, 32, 0, stream>>>(xh, idxb, Wht, bl, nullptr, nullptr,
                                                   s1, q1, nullptr, cpad, h);
        finalize_stats<<<(h + 255) / 256, 256, 0, stream>>>(s1, q1, eg, eb,
                                                            1.0f / (float)(B_ * N_ * KNN),
                                                            sc1, sh1, h);
        // --- EdgeConv pass B: BN1 + ReLU + max over K ---
        edgeconv_wmma<1><<<egrid, 32, 0, stream>>>(xh, idxb, Wht, bl, sc1, sh1,
                                                   nullptr, nullptr, (unsigned int*)xP, cpad, h);

        // --- BN2 + ReLU -> next features (f32 + f16 shadow) ---
        chan_stats<<<h, 256, 0, stream>>>(xP, s2, q2, B_ * N_, h);
        finalize_stats<<<(h + 255) / 256, 256, 0, stream>>>(s2, q2, ng, nb,
                                                            1.0f / (float)(B_ * N_),
                                                            sc2, sh2, h);
        float* xout = outbufs[l & 1];
        {
            int n = B_ * N_ * h;
            bn_apply_f16<<<(n + 255) / 256, 256, 0, stream>>>(xP, sc2, sh2, xout, xh, n, h);
        }
        xcur = xout;
        c = h;
        cpad = h;   // later layers: h is a multiple of 32, no padding needed
    }

    // --- head ---
    const float* fcW1  = (const float*)d_in[25];
    const float* fcb1  = (const float*)d_in[26];
    const float* fcg1  = (const float*)d_in[27];
    const float* fcbb1 = (const float*)d_in[28];
    const float* fcW2  = (const float*)d_in[29];
    const float* fcb2  = (const float*)d_in[30];
    const float* fcg2  = (const float*)d_in[31];
    const float* fcbb2 = (const float*)d_in[32];

    mean_pool<<<(B_ * 512 + 255) / 256, 256, 0, stream>>>(xcur, pooled, 512);
    fc_bn<<<(512 + 255) / 256, 256, 0, stream>>>(pooled, fcW1, fcb1, fcg1, fcbb1, hdd, 512, 512, 1);
    fc_bn<<<(256 + 255) / 256, 256, 0, stream>>>(hdd, fcW2, fcb2, fcg2, fcbb2, (float*)d_out, 512, 256, 0);
}
